// MultiGCN_sigma_63084479644123
// MI455X (gfx1250) — matrix operand order repaired and checked by reading, not compiled, not run on hardware
//
#include <hip/hip_runtime.h>

// ---------------------------------------------------------------------------
// MultiGCN_sigma for MI455X (gfx1250, wave32, WMMA).
// Heavy GEMMs (gram = E E^T, an@an, X W, A (XW)) run on
// v_wmma_f32_16x16x32_bf16 with fp32 accumulation. The K loop is left simple
// (loads feeding WMMAs directly): the compiler unrolls x4 and clauses the
// b128 loads, and latency is hidden by multi-wave occupancy (~60 VGPRs/wave).
// Explicit register double-buffering was tried and spilled to scratch; this
// version keeps the working set small so there are no spills in the hot loop.
// Small relation-CNN, radix-select top-k threshold, and normalization are
// VALU kernels.
// ---------------------------------------------------------------------------

#define NROW 4096
#define DFEA 1600          // 64*5*5
#define HID  1000
#define HPAD 1024          // HID padded to multiple of 64 for WMMA tiling
#define KSEL 819           // round(4096/5)
#define EPSF 2.220446049250313e-16f
#define BN_SCALE 0.9999950000374996f   // 1/sqrt(1+1e-5)

typedef __attribute__((ext_vector_type(16))) __bf16 v16bf;
typedef __attribute__((ext_vector_type(8)))  float  v8f;

union FragU { uint4 u[2]; v16bf v; };

__device__ inline v8f v8f_zero() {
  v8f v;
#pragma unroll
  for (int i = 0; i < 8; ++i) v[i] = 0.0f;
  return v;
}

// A-matrix fragment, 16(M) x 32(K), row-major source with leading dim ld.
// Lanes 0-15 hold K 0..15, lanes 16-31 hold K 16..31 (matches B fragment);
// one 32-byte contiguous run per lane.
__device__ inline v16bf load_fragA(const __bf16* p, int ld, int lane) {
  int m  = lane & 15;
  int kb = (lane >> 4) << 4;              // 0 or 16
  const __bf16* q = p + (size_t)m * ld + kb;
  FragU f;
  f.u[0] = *(const uint4*)(q);            // K = kb .. kb+7
  f.u[1] = *(const uint4*)(q + 8);        // K = kb+8 .. kb+15
  return f.v;
}

// B-matrix fragment, 32(K) x 16(N), sourced from B^T stored row-major [n][k].
// Lanes 0-15 hold K 0..15 of column n, lanes 16-31 hold K 16..31: one
// 32-byte contiguous run per lane.
__device__ inline v16bf load_fragB(const __bf16* p, int ld, int lane) {
  int n  = lane & 15;
  int ko = (lane >> 4) << 4;              // 0 or 16
  const __bf16* q = p + (size_t)n * ld + ko;
  FragU f;
  f.u[0] = *(const uint4*)(q);
  f.u[1] = *(const uint4*)(q + 8);
  return f.v;
}

// One wave computes a 16(M) x 64(N) strip: A-frag reused across 4 B-frags.
__device__ inline void gemm_tile_16x64(const __bf16* A, int ldA,
                                       const __bf16* BT, int ldB,
                                       int row0, int col0, int K, int lane,
                                       v8f acc[4]) {
  const __bf16* arow = A + (size_t)row0 * ldA;
  for (int k0 = 0; k0 < K; k0 += 32) {
    v16bf a = load_fragA(arow + k0, ldA, lane);
#pragma unroll
    for (int j = 0; j < 4; ++j) {
      v16bf b = load_fragB(BT + (size_t)(col0 + 16 * j) * ldB + k0, ldB, lane);
      acc[j] = __builtin_amdgcn_wmma_f32_16x16x32_bf16(
          false, a, false, b, (short)0, acc[j], false, false);
    }
  }
}

// ---------------------------------------------------------------------------
// 1) Relation network: per-sample tiny CNN -> sigma[N]
// ---------------------------------------------------------------------------
__global__ void sigma_kernel(const float* __restrict__ feats,
                             const float* __restrict__ c1w, const float* __restrict__ c1b,
                             const float* __restrict__ g1,  const float* __restrict__ b1,
                             const float* __restrict__ c2w, const float* __restrict__ c2b,
                             const float* __restrict__ g2,  const float* __restrict__ b2,
                             const float* __restrict__ f3w, const float* __restrict__ f3b,
                             const float* __restrict__ f4w, const float* __restrict__ f4b,
                             float* __restrict__ sigma) {
  __shared__ float sfeat[DFEA];
  __shared__ float sbuf1[DFEA];
  __shared__ float spool[576];
  __shared__ float sc2[9];
  __shared__ float sp2[4];
  __shared__ float sh[8];
  const int tid = threadIdx.x;
  const float* f = feats + (size_t)blockIdx.x * DFEA;
  for (int i = tid; i < DFEA; i += 256) sfeat[i] = f[i];
  __syncthreads();
  // conv1 (64->64, 3x3, pad1 on 5x5) + eval-BN + ReLU
  for (int idx = tid; idx < 1600; idx += 256) {
    int oc = idx / 25, r = idx % 25, oy = r / 5, ox = r % 5;
    float acc = c1b[oc];
    for (int ic = 0; ic < 64; ++ic) {
      const float* fin = sfeat + ic * 25;
      const float* w = c1w + ((size_t)oc * 64 + ic) * 9;
      for (int ky = 0; ky < 3; ++ky) {
        int iy = oy + ky - 1;
        if (iy < 0 || iy >= 5) continue;
        for (int kx = 0; kx < 3; ++kx) {
          int ix = ox + kx - 1;
          if (ix < 0 || ix >= 5) continue;
          acc += fin[iy * 5 + ix] * w[ky * 3 + kx];
        }
      }
    }
    acc = acc * (BN_SCALE * g1[oc]) + b1[oc];
    sbuf1[idx] = fmaxf(acc, 0.0f);
  }
  __syncthreads();
  // maxpool 2x2 stride 2 pad 1 (-inf pad; post-ReLU, valid cells suffice): 5x5 -> 3x3
  for (int idx = tid; idx < 576; idx += 256) {
    int oc = idx / 9, r = idx % 9, py = r / 3, px = r % 3;
    float m = -1e30f;
    for (int dy = 0; dy < 2; ++dy) {
      int y = py * 2 - 1 + dy; if (y < 0 || y >= 5) continue;
      for (int dx = 0; dx < 2; ++dx) {
        int x = px * 2 - 1 + dx; if (x < 0 || x >= 5) continue;
        m = fmaxf(m, sbuf1[oc * 25 + y * 5 + x]);
      }
    }
    spool[idx] = m;
  }
  __syncthreads();
  // conv2 (64->1, 3x3, pad1 on 3x3) + BN + ReLU
  if (tid < 9) {
    int oy = tid / 3, ox = tid % 3;
    float acc = c2b[0];
    for (int ic = 0; ic < 64; ++ic) {
      for (int ky = 0; ky < 3; ++ky) {
        int iy = oy + ky - 1; if (iy < 0 || iy >= 3) continue;
        for (int kx = 0; kx < 3; ++kx) {
          int ix = ox + kx - 1; if (ix < 0 || ix >= 3) continue;
          acc += spool[ic * 9 + iy * 3 + ix] * c2w[ic * 9 + ky * 3 + kx];
        }
      }
    }
    acc = acc * (BN_SCALE * g2[0]) + b2[0];
    sc2[tid] = fmaxf(acc, 0.0f);
  }
  __syncthreads();
  // maxpool: 3x3 -> 2x2
  if (tid < 4) {
    int py = tid / 2, px = tid % 2;
    float m = -1e30f;
    for (int dy = 0; dy < 2; ++dy) {
      int y = py * 2 - 1 + dy; if (y < 0 || y >= 3) continue;
      for (int dx = 0; dx < 2; ++dx) {
        int x = px * 2 - 1 + dx; if (x < 0 || x >= 3) continue;
        m = fmaxf(m, sc2[y * 3 + x]);
      }
    }
    sp2[tid] = m;
  }
  __syncthreads();
  if (tid < 8) {
    float acc = f3b[tid];
    for (int i = 0; i < 4; ++i) acc += sp2[i] * f3w[tid * 4 + i];
    sh[tid] = fmaxf(acc, 0.0f);
  }
  __syncthreads();
  if (tid == 0) {
    float acc = f4b[0];
    for (int j = 0; j < 8; ++j) acc += sh[j] * f4w[j];
    sigma[blockIdx.x] = acc;
  }
}

// ---------------------------------------------------------------------------
// 2) Elementwise prep kernels
// ---------------------------------------------------------------------------
__global__ void conv_feats_bf16(const float* __restrict__ x, __bf16* __restrict__ y) {
  size_t total = (size_t)NROW * DFEA;
  for (size_t i = (size_t)blockIdx.x * blockDim.x + threadIdx.x; i < total;
       i += (size_t)gridDim.x * blockDim.x)
    y[i] = (__bf16)x[i];
}

// gcn_w [1600,1000] -> transposed, zero-padded bf16 [1024,1600]
__global__ void conv_gcnw_bf16(const float* __restrict__ w, __bf16* __restrict__ wt) {
  size_t total = (size_t)HPAD * DFEA;
  for (size_t i = (size_t)blockIdx.x * blockDim.x + threadIdx.x; i < total;
       i += (size_t)gridDim.x * blockDim.x) {
    int n = (int)(i / DFEA), k = (int)(i % DFEA);
    float v = (n < HID) ? w[(size_t)k * HID + n] : 0.0f;
    wt[i] = (__bf16)v;
  }
}

// e = features/(sigma+eps) (bf16 for WMMA) and sq[i] = sum_d e^2 (fp32)
__global__ void make_e_kernel(const float* __restrict__ feats,
                              const float* __restrict__ sigma,
                              __bf16* __restrict__ E, float* __restrict__ sq) {
  __shared__ float red[256];
  int tid = threadIdx.x, row = blockIdx.x;
  float s = 1.0f / (sigma[row] + EPSF);
  const float* f = feats + (size_t)row * DFEA;
  __bf16* e = E + (size_t)row * DFEA;
  float acc = 0.0f;
  for (int d = tid; d < DFEA; d += 256) {
    float v = f[d] * s;
    e[d] = (__bf16)v;
    acc += v * v;
  }
  red[tid] = acc;
  __syncthreads();
  for (int st = 128; st > 0; st >>= 1) {
    if (tid < st) red[tid] += red[tid + st];
    __syncthreads();
  }
  if (tid == 0) sq[row] = red[0];
}

// ---------------------------------------------------------------------------
// 3) Gram WMMA GEMM with fused adjacency-kernel epilogue:
//    adj_e = exp(-0.5 * (sq_i + sq_j - 2 e_i.e_j)/D)
// ---------------------------------------------------------------------------
__global__ void gram_adj_kernel(const __bf16* __restrict__ E,
                                const float* __restrict__ sq,
                                float* __restrict__ adj) {
  int lane = threadIdx.x & 31, wave = threadIdx.x >> 5;
  int row0 = blockIdx.y * 16;
  int col0 = (blockIdx.x * 8 + wave) * 64;
  v8f acc[4];
#pragma unroll
  for (int j = 0; j < 4; ++j) acc[j] = v8f_zero();
  gemm_tile_16x64(E, DFEA, E, DFEA, row0, col0, DFEA, lane, acc);
  int n = lane & 15, hi = lane >> 4;
#pragma unroll
  for (int j = 0; j < 4; ++j) {
#pragma unroll
    for (int r = 0; r < 8; ++r) {
      int gi = row0 + r + hi * 8;
      int gj = col0 + 16 * j + n;
      float W = (sq[gi] + sq[gj] - 2.0f * acc[j][r]) * (1.0f / DFEA);
      adj[(size_t)gi * NROW + gj] = __expf(-0.5f * W);
    }
  }
}

// ---------------------------------------------------------------------------
// 4) Exact top-k threshold per row: 4-pass radix select on positive-float
//    bit patterns (monotone as u32). thresh[i] = k-th largest value of row i.
// ---------------------------------------------------------------------------
__global__ void topk_thresh_kernel(const float* __restrict__ adj,
                                   float* __restrict__ thresh) {
  __shared__ unsigned hist[256];
  __shared__ unsigned s_prefix, s_mask;
  __shared__ int s_k;
  int tid = threadIdx.x;
  const unsigned* row = (const unsigned*)(adj + (size_t)blockIdx.x * NROW);
  if (tid == 0) { s_prefix = 0u; s_mask = 0u; s_k = KSEL; }
  __syncthreads();
  for (int pass = 0; pass < 4; ++pass) {
    int shift = 24 - 8 * pass;
    hist[tid] = 0u;
    __syncthreads();
    unsigned pre = s_prefix, msk = s_mask;
    for (int j = tid; j < NROW; j += 256) {
      unsigned u = row[j];
      if ((u & msk) == pre) atomicAdd(&hist[(u >> shift) & 255u], 1u);
    }
    __syncthreads();
    if (tid == 0) {
      int kk = s_k;
      unsigned cum = 0;
      int sel = 0;
      for (int b = 255; b >= 0; --b) {
        unsigned c = hist[b];
        if (cum + c >= (unsigned)kk) { sel = b; s_k = kk - (int)cum; break; }
        cum += c;
      }
      s_prefix = pre | ((unsigned)sel << shift);
      s_mask = msk | (0xFFu << shift);
    }
    __syncthreads();
  }
  if (tid == 0) thresh[blockIdx.x] = __uint_as_float(s_prefix);
}

// d_i = sqrt(rowsum(adjacency)+1); store 1/d_i
__global__ void rowsum_dinv_kernel(const float* __restrict__ adj,
                                   const float* __restrict__ thresh,
                                   float* __restrict__ dinv) {
  __shared__ float red[256];
  int tid = threadIdx.x, i = blockIdx.x;
  float T = thresh[i];
  const float* row = adj + (size_t)i * NROW;
  float acc = 0.0f;
  for (int j = tid; j < NROW; j += 256) {
    float a = row[j];
    acc += (j == i) ? 1.0f : ((a >= T) ? a : 0.0f);
  }
  red[tid] = acc;
  __syncthreads();
  for (int st = 128; st > 0; st >>= 1) {
    if (tid < st) red[tid] += red[tid + st];
    __syncthreads();
  }
  if (tid == 0) dinv[i] = rsqrtf(red[0] + 1.0f);
}

// an = adjacency * dinv_i * dinv_j, stored bf16 both row-major and transposed
__global__ void make_an_kernel(const float* __restrict__ adj,
                               const float* __restrict__ thresh,
                               const float* __restrict__ dinv,
                               __bf16* __restrict__ AN, __bf16* __restrict__ ANT) {
  size_t total = (size_t)NROW * NROW;
  for (size_t idx = (size_t)blockIdx.x * blockDim.x + threadIdx.x; idx < total;
       idx += (size_t)gridDim.x * blockDim.x) {
    int i = (int)(idx >> 12), j = (int)(idx & 4095);
    float a = adj[idx];
    float v = (i == j) ? 1.0f : ((a >= thresh[i]) ? a : 0.0f);
    v *= dinv[i] * dinv[j];
    __bf16 b = (__bf16)v;
    AN[idx] = b;
    ANT[(size_t)j * NROW + i] = b;
  }
}

// ---------------------------------------------------------------------------
// 5) an@an WMMA GEMM fused with aifa mixing: A = f0*I + f1*an + f2*an^2 (bf16)
// ---------------------------------------------------------------------------
__global__ void an2_mix_kernel(const __bf16* __restrict__ AN,
                               const __bf16* __restrict__ ANT,
                               const float* __restrict__ aifa2p,
                               const float* __restrict__ aifa3p,
                               __bf16* __restrict__ Amix) {
  int lane = threadIdx.x & 31, wave = threadIdx.x >> 5;
  int row0 = blockIdx.y * 16;
  int col0 = (blockIdx.x * 8 + wave) * 64;
  v8f acc[4];
#pragma unroll
  for (int j = 0; j < 4; ++j) acc[j] = v8f_zero();
  gemm_tile_16x64(AN, NROW, ANT, NROW, row0, col0, NROW, lane, acc);
  float a2 = aifa2p[0], a3 = aifa3p[0];
  float m = fmaxf(0.0f, fmaxf(a2, a3));
  float e0 = __expf(0.0f - m), e1 = __expf(a2 - m), e2 = __expf(a3 - m);
  float inv = 1.0f / (e0 + e1 + e2);
  float f0 = e0 * inv, f1 = e1 * inv, f2 = e2 * inv;
  int n = lane & 15, hi = lane >> 4;
#pragma unroll
  for (int j = 0; j < 4; ++j) {
#pragma unroll
    for (int r = 0; r < 8; ++r) {
      int gi = row0 + r + hi * 8;
      int gj = col0 + 16 * j + n;
      size_t idx = (size_t)gi * NROW + gj;
      float v = f1 * (float)AN[idx] + f2 * acc[j][r] + ((gi == gj) ? f0 : 0.0f);
      Amix[idx] = (__bf16)v;
    }
  }
}

// ---------------------------------------------------------------------------
// 6) XW = features @ gcn_w  (store transposed bf16 [HPAD, NROW])
// ---------------------------------------------------------------------------
__global__ void xw_gemm_kernel(const __bf16* __restrict__ F,
                               const __bf16* __restrict__ WT,
                               __bf16* __restrict__ XWT) {
  int lane = threadIdx.x & 31, wave = threadIdx.x >> 5;
  int row0 = blockIdx.y * 16;
  int col0 = (blockIdx.x * 8 + wave) * 64;
  v8f acc[4];
#pragma unroll
  for (int j = 0; j < 4; ++j) acc[j] = v8f_zero();
  gemm_tile_16x64(F, DFEA, WT, DFEA, row0, col0, DFEA, lane, acc);
  int n = lane & 15, hi = lane >> 4;
#pragma unroll
  for (int j = 0; j < 4; ++j) {
#pragma unroll
    for (int r = 0; r < 8; ++r) {
      int gi = row0 + r + hi * 8;          // sample
      int gj = col0 + 16 * j + n;          // hidden unit (padded)
      XWT[(size_t)gj * NROW + gi] = (__bf16)acc[j][r];
    }
  }
}

// ---------------------------------------------------------------------------
// 7) out = relu(BN(A @ XW + gcn_b))  -> f32 [NROW, HID]
// ---------------------------------------------------------------------------
__global__ void final_gemm_kernel(const __bf16* __restrict__ A,
                                  const __bf16* __restrict__ XWT,
                                  const float* __restrict__ gcn_b,
                                  const float* __restrict__ bn_g,
                                  const float* __restrict__ bn_b,
                                  float* __restrict__ out) {
  int lane = threadIdx.x & 31, wave = threadIdx.x >> 5;
  int row0 = blockIdx.y * 16;
  int col0 = (blockIdx.x * 8 + wave) * 64;
  v8f acc[4];
#pragma unroll
  for (int j = 0; j < 4; ++j) acc[j] = v8f_zero();
  gemm_tile_16x64(A, NROW, XWT, NROW, row0, col0, NROW, lane, acc);
  int n = lane & 15, hi = lane >> 4;
#pragma unroll
  for (int j = 0; j < 4; ++j) {
#pragma unroll
    for (int r = 0; r < 8; ++r) {
      int gi = row0 + r + hi * 8;
      int gj = col0 + 16 * j + n;
      if (gj < HID) {
        float v = (acc[j][r] + gcn_b[gj]) * (BN_SCALE * bn_g[gj]) + bn_b[gj];
        out[(size_t)gi * HID + gj] = fmaxf(v, 0.0f);
      }
    }
  }
}

// ---------------------------------------------------------------------------
// Workspace layout (bytes), 256-aligned. ~159 MB with aliasing:
//  - E (bf16) aliases feats_bf16 (dead after XW GEMM)
//  - A_mix (bf16) aliases adj_e (dead after make_an)
// ---------------------------------------------------------------------------
static constexpr size_t alup(size_t x) { return (x + 255) & ~(size_t)255; }
static constexpr size_t OFF_FEATSB = 0;
static constexpr size_t OFF_GCNWT  = OFF_FEATSB + alup((size_t)NROW * DFEA * 2);
static constexpr size_t OFF_XWT    = OFF_GCNWT  + alup((size_t)HPAD * DFEA * 2);
static constexpr size_t OFF_SIGMA  = OFF_XWT    + alup((size_t)HPAD * NROW * 2);
static constexpr size_t OFF_SQ     = OFF_SIGMA  + alup((size_t)NROW * 4);
static constexpr size_t OFF_THR    = OFF_SQ     + alup((size_t)NROW * 4);
static constexpr size_t OFF_DINV   = OFF_THR    + alup((size_t)NROW * 4);
static constexpr size_t OFF_ADJ    = OFF_DINV   + alup((size_t)NROW * 4);
static constexpr size_t OFF_AN     = OFF_ADJ    + alup((size_t)NROW * NROW * 4);
static constexpr size_t OFF_ANT    = OFF_AN     + alup((size_t)NROW * NROW * 2);
static constexpr size_t OFF_E      = OFF_FEATSB;   // alias
static constexpr size_t OFF_AMIX   = OFF_ADJ;      // alias

extern "C" void kernel_launch(void* const* d_in, const int* in_sizes, int n_in,
                              void* d_out, int out_size, void* d_ws, size_t ws_size,
                              hipStream_t stream) {
  (void)in_sizes; (void)n_in; (void)out_size; (void)ws_size;
  const float* features = (const float*)d_in[0];
  const float* conv1_w  = (const float*)d_in[1];
  const float* conv1_b  = (const float*)d_in[2];
  const float* bn1_g    = (const float*)d_in[3];
  const float* bn1_b    = (const float*)d_in[4];
  const float* conv2_w  = (const float*)d_in[5];
  const float* conv2_b  = (const float*)d_in[6];
  const float* bn2c_g   = (const float*)d_in[7];
  const float* bn2c_b   = (const float*)d_in[8];
  const float* fc3_w    = (const float*)d_in[9];
  const float* fc3_b    = (const float*)d_in[10];
  const float* fc4_w    = (const float*)d_in[11];
  const float* fc4_b    = (const float*)d_in[12];
  const float* aifa2    = (const float*)d_in[13];
  const float* aifa3    = (const float*)d_in[14];
  const float* gcn_w    = (const float*)d_in[15];
  const float* gcn_b    = (const float*)d_in[16];
  const float* bn_g     = (const float*)d_in[17];
  const float* bn_b     = (const float*)d_in[18];
  float* out = (float*)d_out;

  char* ws = (char*)d_ws;
  __bf16* featsB = (__bf16*)(ws + OFF_FEATSB);
  __bf16* gcnwT  = (__bf16*)(ws + OFF_GCNWT);
  __bf16* xwT    = (__bf16*)(ws + OFF_XWT);
  float*  sigma  = (float*) (ws + OFF_SIGMA);
  float*  sq     = (float*) (ws + OFF_SQ);
  float*  thresh = (float*) (ws + OFF_THR);
  float*  dinv   = (float*) (ws + OFF_DINV);
  float*  adjE   = (float*) (ws + OFF_ADJ);
  __bf16* anB    = (__bf16*)(ws + OFF_AN);
  __bf16* anTB   = (__bf16*)(ws + OFF_ANT);
  __bf16* eB     = (__bf16*)(ws + OFF_E);
  __bf16* amixB  = (__bf16*)(ws + OFF_AMIX);

  // 1) bf16 prep + XW GEMM (uses featsB before E aliases it)
  conv_feats_bf16<<<1024, 256, 0, stream>>>(features, featsB);
  conv_gcnw_bf16<<<1024, 256, 0, stream>>>(gcn_w, gcnwT);
  xw_gemm_kernel<<<dim3(2, 256), 256, 0, stream>>>(featsB, gcnwT, xwT);

  // 2) relation network -> sigma
  sigma_kernel<<<NROW, 256, 0, stream>>>(features, conv1_w, conv1_b, bn1_g, bn1_b,
                                         conv2_w, conv2_b, bn2c_g, bn2c_b,
                                         fc3_w, fc3_b, fc4_w, fc4_b, sigma);

  // 3) e, sq, gram -> adj_e
  make_e_kernel<<<NROW, 256, 0, stream>>>(features, sigma, eB, sq);
  gram_adj_kernel<<<dim3(8, 256), 256, 0, stream>>>(eB, sq, adjE);

  // 4) per-row top-k threshold, degree normalization, an / an^T (bf16)
  topk_thresh_kernel<<<NROW, 256, 0, stream>>>(adjE, thresh);
  rowsum_dinv_kernel<<<NROW, 256, 0, stream>>>(adjE, thresh, dinv);
  make_an_kernel<<<4096, 256, 0, stream>>>(adjE, thresh, dinv, anB, anTB);

  // 5) A = f0*I + f1*an + f2*(an@an)   (A_mix aliases adj_e, now dead)
  an2_mix_kernel<<<dim3(8, 256), 256, 0, stream>>>(anB, anTB, aifa2, aifa3, amixB);

  // 6) out = relu(BN(A @ XW + b))
  final_gemm_kernel<<<dim3(2, 256), 256, 0, stream>>>(amixB, xwT, gcn_b, bn_g, bn_b, out);
}
